// SetApstractionLayer_84808424227388
// MI455X (gfx1250) — matrix operand order repaired and checked by reading, not compiled, not run on hardware
//
#include <hip/hip_runtime.h>

typedef __attribute__((ext_vector_type(16))) _Float16 v16h;
typedef __attribute__((ext_vector_type(8)))  _Float16 v8h;
typedef __attribute__((ext_vector_type(8)))  float    v8f;

#if defined(__AMDGCN__) && __has_builtin(__builtin_amdgcn_tensor_load_to_lds)
#define HAS_TDM 1
typedef unsigned int tdm_v4u __attribute__((ext_vector_type(4)));
typedef int tdm_v4i __attribute__((ext_vector_type(4)));
typedef int tdm_v8i __attribute__((ext_vector_type(8)));
#else
#define HAS_TDM 0
#endif

#define BB 4
#define NN 4096
#define FF 6
#define NC 1024
#define NS 64
#define PP (NC * NS) /* 65536 */
#define RAD2 0.04f

// d_out layout (floats): centroids(B,1024,3) | g_xyz(B,3,1024,64) | feats(B,256,1024) | inT(B,6,6) | feT(B,64,64)
#define OFF_CENT  0
#define OFF_GXYZ  12288
#define OFF_FEATS 798720
#define OFF_INT   1847296
#define OFF_FET   1847440

// ---------------------------------------------------------------------------
// WMMA GEMM, operand-swapped so every LDS access is a contiguous 16B vector:
//   C(MxN) = act( Act(MxK) @ W(KxN) ),  weights staged transposed Wt(NxK).
//   Per 16x16 tile we compute D = Wt_tile x Act_tile^T = C^T tile; the D VGPR
//   layout then maps each lane to one activation row m and 8 consecutive n,
//   so the store is a single ds_store_b128 and C stays row-major MxN.
// ---------------------------------------------------------------------------
__device__ __forceinline__ v8f wmma_tile_T(const _Float16* __restrict__ Wr,
                                           const _Float16* __restrict__ Ar, int K,
                                           int lh) {
  v8f acc = {};
  for (int k0 = 0; k0 < K; k0 += 32) {
    v16h av, bv;
    // A-frag (Wt row): lanes 0-15 K=k0..k0+15, 16-31 K=+16
    const _Float16* a0 = Wr + k0 + lh * 16;
#pragma unroll
    for (int j = 0; j < 8; ++j) av[j] = a0[j];
#pragma unroll
    for (int j = 0; j < 8; ++j) av[8 + j] = a0[8 + j];
    // B-frag (Act row as transposed-B): lanes 0-15 K=k0..k0+15, 16-31 K=+16
    const _Float16* b0 = Ar + k0 + lh * 16;
#pragma unroll
    for (int j = 0; j < 16; ++j) bv[j] = b0[j];
    acc = __builtin_amdgcn_wmma_f32_16x16x32_f16(false, av, false, bv, (short)0, acc,
                                                 false, false);
  }
  return acc;
}

__device__ __forceinline__ void gemm_lds(const _Float16* Act, const _Float16* Wt,
                                         _Float16* C, int M, int N, int K, bool relu,
                                         int nWaves) {
  const int wave = threadIdx.x >> 5;
  const int lane = threadIdx.x & 31;
  const int lh = lane >> 4, r = lane & 15;
  const int mT = M >> 4, nT = N >> 4;
  for (int t = wave; t < mT * nT; t += nWaves) {
    const int mt = t / nT, nt = t % nT;
    v8f acc = wmma_tile_T(Wt + (nt * 16 + r) * K, Act + (mt * 16 + r) * K, K, lh);
    const int m = mt * 16 + r;
    const int nb = nt * 16 + lh * 8;
    v8h o;
#pragma unroll
    for (int i = 0; i < 8; ++i) {
      float v = acc[i];
      if (relu) v = fmaxf(v, 0.f);
      o[i] = (_Float16)v;
    }
    *(v8h*)(C + m * N + nb) = o;   // one 16B LDS store per tile per lane
  }
}

// colmax[n] = max over rows of relu(Act@W); relu implicit (seed 0)
__device__ __forceinline__ void gemm_colmax(const _Float16* Act, const _Float16* Wt,
                                            float* colmax, int M, int N, int K,
                                            int nWaves) {
  const int wave = threadIdx.x >> 5;
  const int lane = threadIdx.x & 31;
  const int lh = lane >> 4, r = lane & 15;
  const int mT = M >> 4, nT = N >> 4;
  for (int t = wave; t < mT * nT; t += nWaves) {
    const int mt = t / nT, nt = t % nT;
    v8f acc = wmma_tile_T(Wt + (nt * 16 + r) * K, Act + (mt * 16 + r) * K, K, lh);
    const int nb = nt * 16 + lh * 8;
#pragma unroll
    for (int i = 0; i < 8; ++i)
      atomicMax((int*)&colmax[nb + i], __float_as_int(fmaxf(acc[i], 0.f)));
  }
}

// Stage W(KxN) from global f32 into LDS transposed as Wt(NxK) f16.
__device__ __forceinline__ void load_wT_f16(const float* __restrict__ W,
                                            _Float16* __restrict__ L, int K, int N,
                                            int nthreads) {
  for (int i = threadIdx.x; i < K * N; i += nthreads) {
    __builtin_prefetch(W + i + 4 * nthreads, 0, 1);  // global_prefetch_b8
    const int k = i / N, n = i % N;                  // coalesced global read
    L[n * K + k] = (_Float16)W[i];
  }
}

// ---------------------------------------------------------------------------
// K0: zero the tnet max accumulators (ws is poisoned each run)
// ---------------------------------------------------------------------------
__global__ void init_kernel(float* tmax1, float* tmax2, int n) {
  int i = blockIdx.x * 256 + threadIdx.x;
  if (i < n) { tmax1[i] = 0.f; tmax2[i] = 0.f; }
}

// ---------------------------------------------------------------------------
// K1: farthest point sampling. 1 block per batch; coords + dist in LDS.
// ---------------------------------------------------------------------------
__global__ void fps_kernel(const float* __restrict__ points, int* __restrict__ idx,
                           float* __restrict__ out) {
  const int b = blockIdx.x;
  extern __shared__ char smem[];
  float* px = (float*)smem;
  float* py = px + NN;
  float* pz = py + NN;
  float* dist = pz + NN;
  float* rmax = dist + NN;
  int*   rarg = (int*)(rmax + 256);
  const float* P = points + (size_t)b * NN * 3;
  for (int i = threadIdx.x; i < NN; i += 256) {
    px[i] = P[3 * i]; py[i] = P[3 * i + 1]; pz[i] = P[3 * i + 2];
    dist[i] = 1e9f;
  }
  __syncthreads();
  int last = 0;
  if (threadIdx.x == 0) {
    idx[b * NC] = 0;
    out[OFF_CENT + (size_t)(b * NC) * 3 + 0] = px[0];
    out[OFF_CENT + (size_t)(b * NC) * 3 + 1] = py[0];
    out[OFF_CENT + (size_t)(b * NC) * 3 + 2] = pz[0];
  }
  for (int it = 1; it < NC; ++it) {
    const float cx = px[last], cy = py[last], cz = pz[last];
    __syncthreads();
    float bd = -1.f; int bi = 0;
    for (int i = threadIdx.x; i < NN; i += 256) {
      float dx = px[i] - cx, dy = py[i] - cy, dz = pz[i] - cz;
      float nd = fminf(dist[i], dx * dx + dy * dy + dz * dz);
      dist[i] = nd;
      if (nd > bd) { bd = nd; bi = i; }
    }
    rmax[threadIdx.x] = bd; rarg[threadIdx.x] = bi;
    __syncthreads();
    for (int s = 128; s > 0; s >>= 1) {
      if (threadIdx.x < s && rmax[threadIdx.x + s] > rmax[threadIdx.x]) {
        rmax[threadIdx.x] = rmax[threadIdx.x + s];
        rarg[threadIdx.x] = rarg[threadIdx.x + s];
      }
      __syncthreads();
    }
    last = rarg[0];
    if (threadIdx.x == 0) {
      idx[b * NC + it] = last;
      out[OFF_CENT + (size_t)(b * NC + it) * 3 + 0] = px[last];
      out[OFF_CENT + (size_t)(b * NC + it) * 3 + 1] = py[last];
      out[OFF_CENT + (size_t)(b * NC + it) * 3 + 2] = pz[last];
    }
  }
}

// ---------------------------------------------------------------------------
// K2: ball query + gather g_xyz (transposed out) + g_feat (f16, padded to 8)
// ---------------------------------------------------------------------------
__global__ void ballquery_kernel(const float* __restrict__ points,
                                 const float* __restrict__ x,
                                 const int* __restrict__ idx, int* __restrict__ gidx,
                                 float* __restrict__ out, _Float16* __restrict__ gfeat) {
  const int b = blockIdx.x >> 2;
  const int c = ((blockIdx.x & 3) << 8) + threadIdx.x;
  extern __shared__ char smem[];
  float* px = (float*)smem; float* py = px + NN; float* pz = py + NN;
  const float* P = points + (size_t)b * NN * 3;
  for (int i = threadIdx.x; i < NN; i += 256) {
    px[i] = P[3 * i]; py[i] = P[3 * i + 1]; pz[i] = P[3 * i + 2];
  }
  __syncthreads();
  const int ci = idx[b * NC + c];
  const float cx = px[ci], cy = py[ci], cz = pz[ci];
  const int base = (b * NC + c) * NS;
  int cnt = 0, first = 0;
  for (int i = 0; i < NN && cnt < NS; ++i) {
    float dx = px[i] - cx, dy = py[i] - cy, dz = pz[i] - cz;
    if (dx * dx + dy * dy + dz * dz <= RAD2) {
      if (cnt == 0) first = i;
      gidx[base + cnt] = i;
      ++cnt;
    }
  }
  for (int j = cnt; j < NS; ++j) gidx[base + j] = first;
  const float* X = x + (size_t)b * NN * FF;
  for (int j = 0; j < NS; ++j) {
    int i = gidx[base + j];
    size_t o = OFF_GXYZ + (((size_t)b * 3) * NC + c) * NS + j;
    out[o]                       = px[i] - cx;
    out[o + (size_t)NC * NS]     = py[i] - cy;
    out[o + 2 * (size_t)NC * NS] = pz[i] - cz;
    _Float16* gr = gfeat + ((size_t)b * PP + c * NS + j) * 8;
    for (int f = 0; f < FF; ++f) gr[f] = (_Float16)X[i * FF + f];
    gr[6] = (_Float16)0.f; gr[7] = (_Float16)0.f;
  }
}

// ---------------------------------------------------------------------------
// K3: tnet1 fused chain per 128-row tile: 6->64 (VALU), 64->128, 128->256
// ---------------------------------------------------------------------------
__global__ void tnet1_kernel(const _Float16* __restrict__ gfeat,
                             const float* __restrict__ w1, const float* __restrict__ w2,
                             const float* __restrict__ w3, float* __restrict__ tmax) {
  const int b  = blockIdx.x >> 9;
  const int tb = blockIdx.x & 511;
  extern __shared__ char smem[];
  _Float16* h1  = (_Float16*)smem;          // 128*64
  _Float16* h2  = h1 + 128 * 64;            // 128*128
  _Float16* w2t = h2 + 128 * 128;           // 128x64  (w2^T)
  _Float16* w3t = w2t + 64 * 128;           // 256x128 (w3^T)
  float* colmax = (float*)(w3t + 128 * 256);
  load_wT_f16(w2, w2t, 64, 128, 256);
  load_wT_f16(w3, w3t, 128, 256, 256);
  if (threadIdx.x < 256) colmax[threadIdx.x] = 0.f;
  const _Float16* gf = gfeat + ((size_t)b * PP + tb * 128) * 8;
  for (int e = threadIdx.x; e < 128 * 64; e += 256) {
    int rrow = e >> 6, col = e & 63;
    float s = 0.f;
    for (int f = 0; f < FF; ++f) s += (float)gf[rrow * 8 + f] * w1[f * 64 + col];
    h1[rrow * 64 + col] = (_Float16)fmaxf(s, 0.f);
  }
  __syncthreads();
  gemm_lds(h1, w2t, h2, 128, 128, 64, true, 8);
  __syncthreads();
  gemm_colmax(h2, w3t, colmax, 128, 256, 128, 8);
  __syncthreads();
  if (threadIdx.x < 256)
    atomicMax((int*)&tmax[b * 256 + threadIdx.x], __float_as_int(colmax[threadIdx.x]));
}

// ---------------------------------------------------------------------------
// K4/K6: tnet tail: relu(tmax @ f1) (256->128), @ f2 (128->outDim) + eye
// ---------------------------------------------------------------------------
__global__ void tnet_tail_kernel(const float* __restrict__ tmax,
                                 const float* __restrict__ f1,
                                 const float* __restrict__ f2, float* __restrict__ outp,
                                 int outDim, int eyeStep) {
  const int b = blockIdx.x;
  __shared__ float t[128];
  const float* m = tmax + b * 256;
  if (threadIdx.x < 128) {
    float s = 0.f;
    for (int k = 0; k < 256; ++k) s += m[k] * f1[k * 128 + threadIdx.x];
    t[threadIdx.x] = fmaxf(s, 0.f);
  }
  __syncthreads();
  for (int o = threadIdx.x; o < outDim; o += 256) {
    float s = 0.f;
    for (int k = 0; k < 128; ++k) s += t[k] * f2[k * outDim + o];
    if (o % eyeStep == 0) s += 1.f;
    outp[b * outDim + o] = s;
  }
}

// ---------------------------------------------------------------------------
// K5: stage C per 128-row tile: h6 = flat@inT, h64 = relu(h6@w1) (VALU, spilled
// to ws as f16), then tnet2 chain 64->64->128->256 (WMMA) + colmax
// ---------------------------------------------------------------------------
__global__ void stagec_kernel(const _Float16* __restrict__ gfeat,
                              const float* __restrict__ out_inT,
                              const float* __restrict__ w1, const float* __restrict__ t2w1,
                              const float* __restrict__ t2w2, const float* __restrict__ t2w3,
                              _Float16* __restrict__ h64, float* __restrict__ tmax) {
  const int b  = blockIdx.x >> 9;
  const int tb = blockIdx.x & 511;
  extern __shared__ char smem[];
  _Float16* h6   = (_Float16*)smem;         // 128*8
  _Float16* h64l = h6 + 128 * 8;            // 128*64
  _Float16* u1   = h64l + 128 * 64;         // 128*64
  _Float16* u2   = u1 + 128 * 64;           // 128*128
  _Float16* w1t  = u2 + 128 * 128;          // 64x64   (t2w1^T)
  _Float16* w2t  = w1t + 64 * 64;           // 128x64  (t2w2^T)
  _Float16* w3t  = w2t + 64 * 128;          // 256x128 (t2w3^T)
  float* colmax  = (float*)(w3t + 128 * 256);
  load_wT_f16(t2w1, w1t, 64, 64, 256);
  load_wT_f16(t2w2, w2t, 64, 128, 256);
  load_wT_f16(t2w3, w3t, 128, 256, 256);
  if (threadIdx.x < 256) colmax[threadIdx.x] = 0.f;
  const float* T = out_inT + b * 36;
  const _Float16* gf = gfeat + ((size_t)b * PP + tb * 128) * 8;
  for (int e = threadIdx.x; e < 128 * 8; e += 256) {
    int rrow = e >> 3, cc = e & 7;
    float s = 0.f;
    if (cc < FF)
      for (int f = 0; f < FF; ++f) s += (float)gf[rrow * 8 + f] * T[f * FF + cc];
    h6[e] = (_Float16)s;
  }
  __syncthreads();
  _Float16* hg = h64 + ((size_t)b * PP + tb * 128) * 64;
  for (int e = threadIdx.x; e < 128 * 64; e += 256) {
    int rrow = e >> 6, col = e & 63;
    float s = 0.f;
    for (int f = 0; f < FF; ++f) s += (float)h6[rrow * 8 + f] * w1[f * 64 + col];
    _Float16 v = (_Float16)fmaxf(s, 0.f);
    h64l[e] = v;
    hg[e] = v;
  }
  __syncthreads();
  gemm_lds(h64l, w1t, u1, 128, 64, 64, true, 8);
  __syncthreads();
  gemm_lds(u1, w2t, u2, 128, 128, 64, true, 8);
  __syncthreads();
  gemm_colmax(u2, w3t, colmax, 128, 256, 128, 8);
  __syncthreads();
  if (threadIdx.x < 256)
    atomicMax((int*)&tmax[b * 256 + threadIdx.x], __float_as_int(colmax[threadIdx.x]));
}

// ---------------------------------------------------------------------------
// K7: stage E per centroid (64 rows): h64@feT -> relu(@w2) -> relu(@w3) ->
// column max over 64 samples -> feats[b, :, c]. h64 tile pulled in by the TDM.
// ---------------------------------------------------------------------------
__global__ void stagee_kernel(const _Float16* __restrict__ h64,
                              const float* __restrict__ out_feT,
                              const float* __restrict__ w2, const float* __restrict__ w3,
                              float* __restrict__ out) {
  const int b = blockIdx.x >> 10;
  const int c = blockIdx.x & 1023;
  extern __shared__ char smem[];
  _Float16* hl  = (_Float16*)smem;     // 64*64 activations
  _Float16* fTt = hl + 64 * 64;        // 64x64  (feT^T)
  _Float16* g   = fTt + 64 * 64;       // 64*64
  _Float16* w2t = g + 64 * 64;         // 128x64 (w2^T)
  _Float16* a2  = w2t + 64 * 128;      // 64*128
  _Float16* w3t = a2 + 64 * 128;       // 256x128 (w3^T)
  float* colmax = (float*)(w3t + 128 * 256);
  const _Float16* hs = h64 + ((size_t)b * PP + c * NS) * 64;
#if HAS_TDM
  if ((threadIdx.x >> 5) == 0) {  // one TDM DMA per block (EXEC ignored by TDM)
    const unsigned long long ga = (unsigned long long)(const void*)hs;
    const unsigned lds = (unsigned)(unsigned long long)(void*)hl;  // LDS offset
    tdm_v4u g0 = { 1u,                                     // count=1, user D#
                   lds,                                    // lds_addr
                   (unsigned)(ga & 0xFFFFFFFFu),           // global_addr[31:0]
                   (unsigned)(((ga >> 32) & 0x1FFFFFFu) | (2u << 30)) }; // type=2
    tdm_v8i g1;
    g1[0] = (int)(1u << 16);        // data_size = 2 bytes
    g1[1] = (int)(4096u << 16);     // tensor_dim0[15:0] = 4096 (64*64 halfs)
    g1[2] = (int)(1u << 16);        // tensor_dim0 hi=0 | tensor_dim1 lo = 1
    g1[3] = (int)(4096u << 16);     // tensor_dim1 hi=0 | tile_dim0 = 4096
    g1[4] = 1;                      // tile_dim1 = 1, tile_dim2 = 0
    g1[5] = 4096;                   // tensor_dim0_stride = 4096
    g1[6] = 0; g1[7] = 0;
    tdm_v4i gz = {0, 0, 0, 0};
#if __clang_major__ >= 23
    tdm_v8i z8 = {0, 0, 0, 0, 0, 0, 0, 0};
    __builtin_amdgcn_tensor_load_to_lds(g0, g1, gz, gz, z8, 0);
#else
    __builtin_amdgcn_tensor_load_to_lds(g0, g1, gz, gz, 0);
#endif
    __builtin_amdgcn_s_wait_tensorcnt(0);
  }
#else
  for (int i = threadIdx.x; i < 64 * 64; i += 128) hl[i] = hs[i];
#endif
  const float* fT = out_feT + b * 4096;
  for (int i = threadIdx.x; i < 4096; i += 128) {
    int k = i >> 6, n = i & 63;
    fTt[n * 64 + k] = (_Float16)fT[i];   // stage feT transposed
  }
  load_wT_f16(w2, w2t, 64, 128, 128);
  load_wT_f16(w3, w3t, 128, 256, 128);
  for (int i = threadIdx.x; i < 256; i += 128) colmax[i] = 0.f;
  __syncthreads();
  gemm_lds(hl, fTt, g, 64, 64, 64, false, 4);
  __syncthreads();
  gemm_lds(g, w2t, a2, 64, 128, 64, true, 4);
  __syncthreads();
  gemm_colmax(a2, w3t, colmax, 64, 256, 128, 4);
  __syncthreads();
  for (int n = threadIdx.x; n < 256; n += 128)
    out[OFF_FEATS + (size_t)(b * 256 + n) * NC + c] = colmax[n];
}

// ---------------------------------------------------------------------------
extern "C" void kernel_launch(void* const* d_in, const int* in_sizes, int n_in,
                              void* d_out, int out_size, void* d_ws, size_t ws_size,
                              hipStream_t stream) {
  (void)in_sizes; (void)n_in; (void)out_size; (void)ws_size;
  const float* x      = (const float*)d_in[0];
  const float* points = (const float*)d_in[1];
  const float* t1_w1  = (const float*)d_in[2];
  const float* t1_w2  = (const float*)d_in[3];
  const float* t1_w3  = (const float*)d_in[4];
  const float* t1_f1  = (const float*)d_in[5];
  const float* t1_f2  = (const float*)d_in[6];
  const float* t2_w1  = (const float*)d_in[7];
  const float* t2_w2  = (const float*)d_in[8];
  const float* t2_w3  = (const float*)d_in[9];
  const float* t2_f1  = (const float*)d_in[10];
  const float* t2_f2  = (const float*)d_in[11];
  const float* w1     = (const float*)d_in[12];
  const float* w2     = (const float*)d_in[13];
  const float* w3     = (const float*)d_in[14];
  float* out = (float*)d_out;

  char* ws = (char*)d_ws;
  int* idxb = (int*)ws;                                    // B*1024 ints
  int* gidx = (int*)(ws + 16384);                          // B*1024*64 ints
  _Float16* gfeat = (_Float16*)(ws + 16384 + 1048576);     // B*65536*8 f16
  _Float16* h64   = gfeat + (size_t)BB * PP * 8;           // B*65536*64 f16
  float* tmax1 = (float*)(h64 + (size_t)BB * PP * 64);     // B*256
  float* tmax2 = tmax1 + BB * 256;                         // B*256

  const size_t smem_fps = (size_t)NN * 4 * 4 + 256 * 4 + 256 * 4;
  const size_t smem_bq  = (size_t)NN * 3 * 4;
  const size_t smem_t1  = (size_t)(128*64 + 128*128 + 64*128 + 128*256) * 2 + 1024;
  const size_t smem_sc  = (size_t)(128*8 + 128*64 + 128*64 + 128*128 + 64*64 + 64*128 + 128*256) * 2 + 1024;
  const size_t smem_se  = (size_t)(64*64*3 + 64*128*2 + 128*256) * 2 + 1024;

  init_kernel<<<4, 256, 0, stream>>>(tmax1, tmax2, BB * 256);
  fps_kernel<<<BB, 256, smem_fps, stream>>>(points, idxb, out);
  ballquery_kernel<<<BB * 4, 256, smem_bq, stream>>>(points, x, idxb, gidx, out, gfeat);
  tnet1_kernel<<<BB * 512, 256, smem_t1, stream>>>(gfeat, t1_w1, t1_w2, t1_w3, tmax1);
  tnet_tail_kernel<<<BB, 256, 0, stream>>>(tmax1, t1_f1, t1_f2, out + OFF_INT, 36, 7);
  stagec_kernel<<<BB * 512, 256, smem_sc, stream>>>(gfeat, out + OFF_INT, w1, t2_w1,
                                                    t2_w2, t2_w3, h64, tmax2);
  tnet_tail_kernel<<<BB, 256, 0, stream>>>(tmax2, t2_f1, t2_f2, out + OFF_FET, 4096, 65);
  stagee_kernel<<<BB * 1024, 128, smem_se, stream>>>(h64, out + OFF_FET, w2, w3, out);
}
